// RandomSeqWinTransBlock_32899449487878
// MI455X (gfx1250) — compile-verified, matrix operand order checked by reading, not run on hardware
//
#include <hip/hip_runtime.h>

typedef _Float16 half_t;
typedef __attribute__((ext_vector_type(16))) _Float16 v16h;
typedef __attribute__((ext_vector_type(8)))  _Float16 v8h;
typedef __attribute__((ext_vector_type(8)))  float    v8f;

#define BB   8
#define NN   16384
#define CC   128
#define HH   512
#define WIN_ 64
#define NH_  4
#define DH   32
#define ROWS (BB * NN)            // 131072 rows total

// ---------------------------------------------------------------------------
// Fragment loader: 16-bit A-matrix 16x32 layout (ISA 7.12.2).
// lane<16 : row=lane,   K = [0..7] and [16..23]
// lane>=16: row=lane-16,K = [8..15] and [24..31]
// Works for B too, given B^T stored row-major (N x K).
// ---------------------------------------------------------------------------
__device__ __forceinline__ v16h load_frag16(const half_t* __restrict__ p, int ld) {
    int lane = threadIdx.x & 31;
    int row  = lane & 15;
    int k0   = (lane >> 4) << 3;          // 0 or 8
    const half_t* r = p + (size_t)row * ld + k0;
    v8h lo = *(const v8h*)(r);
    v8h hi = *(const v8h*)(r + 16);
    v16h f;
#pragma unroll
    for (int i = 0; i < 8; ++i) { f[i] = lo[i]; f[i + 8] = hi[i]; }
    return f;
}

// ---------------------------------------------------------------------------
// Generic WMMA GEMM:  Out[M,N] = A[M,K] * B[K,N] (B given as B^T, N x K, f16)
// 256 threads = 8 waves; wave tile 64M x 32N; block tile 128M x 128N.
// ---------------------------------------------------------------------------
template <bool RES, bool RELU, bool OUT16>
__global__ __launch_bounds__(256)
void gemm_wmma(const half_t* __restrict__ A, const half_t* __restrict__ BT,
               const float* __restrict__ Res, void* __restrict__ Out,
               int K, int Ntot) {
    const int lane = threadIdx.x & 31;
    const int wave = threadIdx.x >> 5;
    const int wm = wave & 1, wn = wave >> 1;
    const int mbase = blockIdx.x * 128 + wm * 64;
    const int nbase = blockIdx.y * 128 + wn * 32;

    v8f acc[4][2];
#pragma unroll
    for (int mi = 0; mi < 4; ++mi)
#pragma unroll
        for (int ni = 0; ni < 2; ++ni) acc[mi][ni] = {};

    for (int kt = 0; kt < K; kt += 32) {
        if (kt + 32 < K)   // hint next A panel into cache (global_prefetch_b8)
            __builtin_prefetch(A + (size_t)(mbase + (lane & 15)) * K + kt + 32, 0, 1);
        v16h a[4], b[2];
#pragma unroll
        for (int mi = 0; mi < 4; ++mi)
            a[mi] = load_frag16(A + (size_t)(mbase + mi * 16) * K + kt, K);
#pragma unroll
        for (int ni = 0; ni < 2; ++ni)
            b[ni] = load_frag16(BT + (size_t)(nbase + ni * 16) * K + kt, K);
#pragma unroll
        for (int mi = 0; mi < 4; ++mi)
#pragma unroll
            for (int ni = 0; ni < 2; ++ni)
                acc[mi][ni] = __builtin_amdgcn_wmma_f32_16x16x32_f16(
                    false, a[mi], false, b[ni], (short)0, acc[mi][ni], false, false);
    }

    // C/D layout: lanes 0-15 -> M=r, lanes 16-31 -> M=r+8; col = lane&15
    const int col  = lane & 15;
    const int rofs = (lane >> 4) << 3;
#pragma unroll
    for (int mi = 0; mi < 4; ++mi)
#pragma unroll
        for (int ni = 0; ni < 2; ++ni)
#pragma unroll
            for (int r = 0; r < 8; ++r) {
                size_t row = (size_t)mbase + mi * 16 + rofs + r;
                size_t idx = row * (size_t)Ntot + nbase + ni * 16 + col;
                float v = acc[mi][ni][r];
                if (RES)  v += Res[idx];
                if (RELU) v = v > 0.f ? v : 0.f;
                if (OUT16) ((half_t*)Out)[idx] = (half_t)v;
                else       ((float*)Out)[idx]  = v;
            }
}

// ---------------------------------------------------------------------------
// Windowed attention: 1 block = 1 window (64 rows), 1 wave = 1 head (dh=32).
// qkv rows: [q(4x32) | k(4x32) | v(4x32)] = 384 halves.
// ---------------------------------------------------------------------------
__global__ __launch_bounds__(128)
void attn_wmma(const half_t* __restrict__ qkv, half_t* __restrict__ obuf) {
    __shared__ half_t lds[4 * (16 * 64 + 32 * 64)];   // per-wave: P(16x64) + vT(32x64)
    const int lane = threadIdx.x & 31;
    const int head = threadIdx.x >> 5;
    const size_t rowbase = (size_t)blockIdx.x * WIN_;
    const half_t* base = qkv + rowbase * 384;
    const half_t* qp = base + head * DH;
    const half_t* kp = base + 128 + head * DH;
    const half_t* vp = base + 256 + head * DH;
    half_t* P  = lds + head * (16 * 64 + 32 * 64);
    half_t* vT = P + 16 * 64;

    // stage v^T (dh x seq) so O's B-fragments are contiguous; lane = dh row
#pragma unroll 4
    for (int j = 0; j < WIN_; ++j)
        vT[lane * WIN_ + j] = vp[(size_t)j * 384 + lane];

    const float scale = 0.1767766952966369f;   // 1/sqrt(32)
    const int col  = lane & 15;
    const int rofs = (lane >> 4) << 3;

    for (int mt = 0; mt < 4; ++mt) {            // 16-row strip of q
        v16h aq = load_frag16(qp + (size_t)(mt * 16) * 384, 384);
        v8f s[4];
#pragma unroll
        for (int nt = 0; nt < 4; ++nt) {
            v16h bk = load_frag16(kp + (size_t)(nt * 16) * 384, 384);
            v8f z = {};
            s[nt] = __builtin_amdgcn_wmma_f32_16x16x32_f16(
                false, aq, false, bk, (short)0, z, false, false);
        }
        // rowwise softmax: reduce over 4 N-tiles then across the 16-lane half
#pragma unroll
        for (int r = 0; r < 8; ++r) {
            float m = s[0][r];
#pragma unroll
            for (int nt = 1; nt < 4; ++nt) m = fmaxf(m, s[nt][r]);
#pragma unroll
            for (int off = 1; off < 16; off <<= 1) m = fmaxf(m, __shfl_xor(m, off, 32));
            m *= scale;
            float sum = 0.f;
#pragma unroll
            for (int nt = 0; nt < 4; ++nt) {
                float p = __expf(s[nt][r] * scale - m);
                s[nt][r] = p; sum += p;
            }
#pragma unroll
            for (int off = 1; off < 16; off <<= 1) sum += __shfl_xor(sum, off, 32);
            float inv = 1.0f / sum;
            int prow = r + rofs;
#pragma unroll
            for (int nt = 0; nt < 4; ++nt)
                P[prow * 64 + nt * 16 + col] = (half_t)(s[nt][r] * inv);
        }
        // O strip = P(16x64) @ v(64x32)
#pragma unroll
        for (int vt = 0; vt < 2; ++vt) {
            v8f o = {};
#pragma unroll
            for (int kt = 0; kt < 2; ++kt) {
                v16h ap = load_frag16(P + kt * 32, 64);
                v16h bv = load_frag16(vT + (size_t)(vt * 16) * 64 + kt * 32, 64);
                o = __builtin_amdgcn_wmma_f32_16x16x32_f16(
                    false, ap, false, bv, (short)0, o, false, false);
            }
#pragma unroll
            for (int r = 0; r < 8; ++r)
                obuf[(rowbase + mt * 16 + rofs + r) * (size_t)CC +
                     head * DH + vt * 16 + col] = (half_t)o[r];
        }
    }
}

// ---------------------------------------------------------------------------
// BatchNorm helpers (stats are permutation-invariant over (B,N))
// ---------------------------------------------------------------------------
__global__ void zero_stats(float* s) { s[threadIdx.x] = 0.f; }

__global__ __launch_bounds__(256)
void bn_reduce(const float* __restrict__ x, float* __restrict__ sums) {
    __shared__ float ls[2][CC], ls2[2][CC];
    const int c  = threadIdx.x & (CC - 1);
    const int r0 = threadIdx.x >> 7;
    const size_t rowbase = (size_t)blockIdx.x * 1024;
    float s = 0.f, s2 = 0.f;
    for (int r = r0; r < 1024; r += 2) {
        float v = x[(rowbase + r) * CC + c];
        s += v; s2 += v * v;
    }
    ls[r0][c] = s; ls2[r0][c] = s2;
    __syncthreads();
    if (threadIdx.x < CC) {
        atomicAdd(&sums[c],      ls[0][c]  + ls[1][c]);
        atomicAdd(&sums[CC + c], ls2[0][c] + ls2[1][c]);
    }
}

__global__ void bn_finalize(const float* __restrict__ sums,
                            const float* __restrict__ g, const float* __restrict__ b,
                            float* __restrict__ scsh) {
    int c = threadIdx.x;                    // 128 threads
    float mean = sums[c] * (1.0f / ROWS);
    float var  = sums[CC + c] * (1.0f / ROWS) - mean * mean;
    float sc   = g[c] * rsqrtf(var + 1e-5f);
    scsh[c]      = sc;
    scsh[CC + c] = b[c] - mean * sc;
}

// gather-permute + fused normalize (f32 permuted copy + f16 normalized copy)
__global__ __launch_bounds__(256)
void permute_norm(const float* __restrict__ xin, const int* __restrict__ ip,
                  const float* __restrict__ scsh,
                  float* __restrict__ xout, half_t* __restrict__ h) {
    size_t e = (size_t)blockIdx.x * 256 + threadIdx.x;
    int c = e & (CC - 1);
    size_t row = e >> 7, b = row >> 14, n = row & (NN - 1);
    int src = ip[b * NN + n];
    float v = xin[(b * NN + src) * CC + c];
    xout[e] = v;
    h[e] = (half_t)(v * scsh[c] + scsh[CC + c]);
}

__global__ __launch_bounds__(256)
void norm_only(const float* __restrict__ xin, const float* __restrict__ scsh,
               half_t* __restrict__ h) {
    size_t e = (size_t)blockIdx.x * 256 + threadIdx.x;
    int c = e & (CC - 1);
    h[e] = (half_t)(xin[e] * scsh[c] + scsh[CC + c]);
}

// unpermute: since ir = argsort(ip), take(x, ir) == scatter with ip
__global__ __launch_bounds__(256)
void scatter_rows(const float* __restrict__ xin, const int* __restrict__ ip,
                  float* __restrict__ xout) {
    size_t e = (size_t)blockIdx.x * 256 + threadIdx.x;
    int c = e & (CC - 1);
    size_t row = e >> 7, b = row >> 14, n = row & (NN - 1);
    int dst = ip[b * NN + n];
    xout[(b * NN + dst) * CC + c] = xin[e];
}

__global__ __launch_bounds__(256)
void copy_f32(const float* __restrict__ in, float* __restrict__ out, size_t n) {
    size_t i = (size_t)blockIdx.x * 256 + threadIdx.x;
    if (i < n) out[i] = in[i];
}

// transpose-pack weight W(K x Ntot, f32) -> W^T(Ntot x K, f16)
__global__ __launch_bounds__(256)
void packT(const float* __restrict__ W, half_t* __restrict__ WT, int K, int Ntot) {
    size_t e = (size_t)blockIdx.x * 256 + threadIdx.x;
    if (e >= (size_t)K * Ntot) return;
    int n = (int)(e / K), k = (int)(e % K);
    WT[e] = (half_t)W[(size_t)k * Ntot + n];
}

// ---------------------------------------------------------------------------
extern "C" void kernel_launch(void* const* d_in, const int* in_sizes, int n_in,
                              void* d_out, int out_size, void* d_ws, size_t ws_size,
                              hipStream_t stream) {
    const float* x     = (const float*)d_in[0];
    const float* z     = (const float*)d_in[1];
    const int*   ip1   = (const int*)d_in[2];
    const int*   ip2   = (const int*)d_in[3];
    const float* qkv_w = (const float*)d_in[4];
    const float* proj_w= (const float*)d_in[5];
    const float* fc1_w = (const float*)d_in[6];
    const float* fc2_w = (const float*)d_in[7];
    const float* n1_g  = (const float*)d_in[8];
    const float* n1_b  = (const float*)d_in[9];
    const float* n2_g  = (const float*)d_in[10];
    const float* n2_b  = (const float*)d_in[11];

    float* out_x = (float*)d_out;
    float* out_z = out_x + (size_t)ROWS * CC;

    // ---- workspace carve-up ----
    char* p = (char*)d_ws;
    const size_t SZ_X   = (size_t)ROWS * CC * sizeof(float);    // 64 MiB
    const size_t SZ_H   = (size_t)ROWS * CC * sizeof(half_t);   // 32 MiB
    const size_t SZ_QKV = (size_t)ROWS * 384 * sizeof(half_t);  // 96 MiB
    const size_t SZ_ACT = (size_t)ROWS * HH * sizeof(half_t);   // 128 MiB
    float*  xbuf0 = (float*)p;  p += SZ_X;
    float*  xbuf1 = (float*)p;  p += SZ_X;
    half_t* hbuf  = (half_t*)p; p += SZ_H;
    half_t* qkvbuf = (half_t*)p;
    half_t* obuf   = (half_t*)(p + SZ_QKV);   // after qkv
    half_t* actbuf = qkvbuf;                  // aliases qkv+obuf (dead by then)
    p += SZ_ACT;
    half_t* qkvT = (half_t*)p; p += 2 * 384 * CC * sizeof(half_t);
    half_t* projT= (half_t*)p; p += 2 * CC  * CC * sizeof(half_t);
    half_t* fc1T = (half_t*)p; p += 2 * HH  * CC * sizeof(half_t);
    half_t* fc2T = (half_t*)p; p += 2 * CC  * HH * sizeof(half_t);
    float*  stats= (float*)p;  p += 2 * CC * sizeof(float);
    float*  scsh = (float*)p;  p += 2 * CC * sizeof(float);

    const int EW = (ROWS * CC) / 256;   // elementwise grid (65536 blocks)

    // ---- pack weights (once per call, both blocks) ----
    for (int i = 0; i < 2; ++i) {
        packT<<<(CC * 384 + 255) / 256, 256, 0, stream>>>(qkv_w + (size_t)i * CC * 384, qkvT + (size_t)i * 384 * CC, CC, 384);
        packT<<<(CC * CC + 255) / 256, 256, 0, stream>>>(proj_w + (size_t)i * CC * CC,  projT + (size_t)i * CC * CC,  CC, CC);
        packT<<<(CC * HH + 255) / 256, 256, 0, stream>>>(fc1_w + (size_t)i * CC * HH,   fc1T  + (size_t)i * HH * CC,  CC, HH);
        packT<<<(HH * CC + 255) / 256, 256, 0, stream>>>(fc2_w + (size_t)i * HH * CC,   fc2T  + (size_t)i * CC * HH,  HH, CC);
    }
    // z is permuted then un-permuted by its inverse -> identity
    copy_f32<<<(ROWS * 3 + 255) / 256, 256, 0, stream>>>(z, out_z, (size_t)ROWS * 3);

    const float* cur = x;
    for (int i = 0; i < 2; ++i) {
        const int* ip = (i == 0) ? ip1 : ip2;
        // BN1 (stats on un-permuted buffer; permutation-invariant)
        zero_stats<<<1, 256, 0, stream>>>(stats);
        bn_reduce<<<ROWS / 1024, 256, 0, stream>>>(cur, stats);
        bn_finalize<<<1, CC, 0, stream>>>(stats, n1_g + i * CC, n1_b + i * CC, scsh);
        permute_norm<<<EW, 256, 0, stream>>>(cur, ip, scsh, xbuf0, hbuf);
        // QKV: (131072 x 128) @ (128 x 384) -> f16
        gemm_wmma<false, false, true><<<dim3(ROWS / 128, 3), 256, 0, stream>>>(
            hbuf, qkvT + (size_t)i * 384 * CC, nullptr, qkvbuf, CC, 384);
        // windowed attention
        attn_wmma<<<ROWS / WIN_, 128, 0, stream>>>(qkvbuf, obuf);
        // proj + residual(xp) -> f32
        gemm_wmma<true, false, false><<<dim3(ROWS / 128, 1), 256, 0, stream>>>(
            obuf, projT + (size_t)i * CC * CC, xbuf0, xbuf1, CC, CC);
        // BN2
        zero_stats<<<1, 256, 0, stream>>>(stats);
        bn_reduce<<<ROWS / 1024, 256, 0, stream>>>(xbuf1, stats);
        bn_finalize<<<1, CC, 0, stream>>>(stats, n2_g + i * CC, n2_b + i * CC, scsh);
        norm_only<<<EW, 256, 0, stream>>>(xbuf1, scsh, hbuf);
        // FC1 + ReLU -> f16 ; FC2 + residual -> f32
        gemm_wmma<false, true, true><<<dim3(ROWS / 128, 4), 256, 0, stream>>>(
            hbuf, fc1T + (size_t)i * HH * CC, nullptr, actbuf, CC, HH);
        gemm_wmma<true, false, false><<<dim3(ROWS / 128, 1), 256, 0, stream>>>(
            actbuf, fc2T + (size_t)i * CC * HH, xbuf1, xbuf0, HH, CC);
        // un-permute
        float* dst = (i == 0) ? xbuf1 : out_x;
        scatter_rows<<<EW, 256, 0, stream>>>(xbuf0, ip, dst);
        cur = xbuf1;
    }
}
